// SchNetInteraction_19018115186810
// MI455X (gfx1250) — matrix operand order repaired and checked by reading, not compiled, not run on hardware
//
#include <hip/hip_runtime.h>
#include <hip/hip_bf16.h>

// ---------------------------------------------------------------------------
// SchNet interaction block, fused, bf16-WMMA (v_wmma_f32_16x16x32_bf16).
//   h   = x @ W_in2f
//   Wij = (ssp(f_ij @ W_f1 + b_f1) @ W_f2 + b_f2) * rcut
//   agg = segment_sum(h[idx_j] * Wij, idx_i)      (idx_i sorted)
//   out = ssp(agg @ W_o1 + b_o1) @ W_o2 + b_o2
// Weights pre-swizzled once into lane-ordered bf16 B-fragments (2 coalesced
// b128 loads per fragment). f_ij tiles are double-buffered in LDS via the
// CDNA5 async global->LDS path (ASYNCcnt) when available, overlapping the
// next tile's HBM traffic with the current tile's WMMA work.
// ---------------------------------------------------------------------------

#define N_ATOMS 50000
#define N_EDGES 1600000
#define N_BASIS 128
#define N_RBF   20
#define N_FILT  128
#define TILE_E  128        // edges per tile in the edge kernel
#define EDGE_TILES 4       // tiles per workgroup (amortize fragment loads)
#define T_LD    136        // LDS row stride (bf16): 272B -> conflict-free b128

// fragment storage: per 128x128 matrix: 8 n-tiles * 4 k-steps * 32 lanes * 16 bf16
#define FRAG_MAT_ELEMS (8 * 4 * 32 * 16)
#define WS_FRAG_RESERVE (256 * 1024)     // frags live in first 256KB of ws

#define FIJ_TILE_WORDS (TILE_E * N_RBF)          // 2560 dwords = 10240 B
#define FIJ_TILE_VEC4  (FIJ_TILE_WORDS / 4)      // 640 x b128

#if defined(__HIP_DEVICE_COMPILE__) && \
    __has_builtin(__builtin_amdgcn_global_load_async_to_lds_b128)
#define HAVE_ASYNC_LDS 1
#else
#define HAVE_ASYNC_LDS 0
#endif

typedef __attribute__((ext_vector_type(16))) __bf16 v16bf;
typedef __attribute__((ext_vector_type(8)))  float  v8f;
typedef __attribute__((ext_vector_type(4)))  int    v4i;

__device__ __forceinline__ v8f wmma_bf16(v16bf a, v16bf b, v8f c) {
  // (neg_a, A, neg_b, B, c_mod, C, reuse_a, reuse_b)
  return __builtin_amdgcn_wmma_f32_16x16x32_bf16(false, a, false, b, (short)0, c,
                                                 false, false);
}

// shifted softplus: softplus(x) - log(2), numerically stable
__device__ __forceinline__ float ssp(float x) {
  return fmaxf(x, 0.0f) + log1pf(__expf(-fabsf(x))) - 0.6931471805599453f;
}

// A fragment (16x32 bf16) from a row-major fp32 matrix, ld = row stride.
// lane layout: elems 0..7  -> K = k0 + 8*half + i
//              elems 8..15 -> K = k0 + 8*half + 16 + (i-8)
__device__ __forceinline__ v16bf load_a_f32(const float* __restrict__ base, int ld,
                                            int row, int k0, int half) {
  const float* p0 = base + (size_t)row * ld + k0 + 8 * half;
  const float4* q = reinterpret_cast<const float4*>(p0);
  float4 f0 = q[0], f1 = q[1];            // K .. K+7
  const float4* q2 = reinterpret_cast<const float4*>(p0 + 16);
  float4 f2 = q2[0], f3 = q2[1];          // K+16 .. K+23
  v16bf a;
  a[0] = (__bf16)f0.x; a[1] = (__bf16)f0.y; a[2] = (__bf16)f0.z; a[3] = (__bf16)f0.w;
  a[4] = (__bf16)f1.x; a[5] = (__bf16)f1.y; a[6] = (__bf16)f1.z; a[7] = (__bf16)f1.w;
  a[8] = (__bf16)f2.x; a[9] = (__bf16)f2.y; a[10] = (__bf16)f2.z; a[11] = (__bf16)f2.w;
  a[12] = (__bf16)f3.x; a[13] = (__bf16)f3.y; a[14] = (__bf16)f3.z; a[15] = (__bf16)f3.w;
  return a;
}

// A fragment from bf16 LDS tile (ld in bf16 elements).
__device__ __forceinline__ v16bf load_a_lds(const __bf16* T, int ld, int row,
                                            int k0, int half) {
  const __bf16* p0 = T + row * ld + k0 + 8 * half;
  union { v16bf v; uint4 q[2]; } u;
  u.q[0] = *reinterpret_cast<const uint4*>(p0);
  u.q[1] = *reinterpret_cast<const uint4*>(p0 + 16);
  return u.v;
}

// B fragment (32x16 bf16) from row-major fp32 weight [K][N], ld = N (prep only).
__device__ __forceinline__ v16bf load_b_f32(const float* __restrict__ W, int ld,
                                            int k0, int col, int half) {
  const float* p = W + (size_t)(k0 + 8 * half) * ld + col;
  v16bf b;
#pragma unroll
  for (int i = 0; i < 16; ++i) b[i] = (__bf16)p[(size_t)(i + (i >= 8 ? 8 : 0)) * ld];
  return b;
}

// Pre-swizzled fragment fetch: 2 coalesced b128 loads per lane.
__device__ __forceinline__ v16bf load_frag(const __bf16* __restrict__ F, int nt,
                                           int kk, int lane) {
  const uint4* p =
      reinterpret_cast<const uint4*>(F + ((size_t)(nt * 4 + kk) * 32 + lane) * 16);
  union { v16bf v; uint4 q[2]; } u;
  u.q[0] = p[0];
  u.q[1] = p[1];
  return u.v;
}

__device__ __forceinline__ void store_frag(__bf16* F, size_t fragIdx, int lane,
                                           v16bf b) {
  union { v16bf v; uint4 q[2]; } u;
  u.v = b;
  uint4* p = reinterpret_cast<uint4*>(F + (fragIdx * 32 + lane) * 16);
  p[0] = u.q[0];
  p[1] = u.q[1];
}

// Cooperative copy of one f_ij tile (10240 B) into LDS.
// Async path: GLOBAL_LOAD_ASYNC_TO_LDS_B128 (ASYNCcnt), else plain b128 copy.
__device__ __forceinline__ void fij_tile_copy(const float* __restrict__ gsrc,
                                              float* lds_dst, int tid) {
#if HAVE_ASYNC_LDS
  typedef __attribute__((address_space(1))) v4i gv4i_t;
  typedef __attribute__((address_space(3))) v4i lv4i_t;
  gv4i_t* g = (gv4i_t*)(const void*)gsrc;
  lv4i_t* l = (lv4i_t*)(void*)lds_dst;
  for (int i = tid; i < FIJ_TILE_VEC4; i += 256)
    __builtin_amdgcn_global_load_async_to_lds_b128(g + i, l + i, 0, 0);
#else
  const float4* g = reinterpret_cast<const float4*>(gsrc);
  float4* l = reinterpret_cast<float4*>(lds_dst);
  for (int i = tid; i < FIJ_TILE_VEC4; i += 256) l[i] = g[i];
#endif
}

__device__ __forceinline__ void fij_copy_wait() {
#if HAVE_ASYNC_LDS
#if __has_builtin(__builtin_amdgcn_s_wait_asynccnt)
  __builtin_amdgcn_s_wait_asynccnt(0);
#else
  asm volatile("s_wait_asynccnt 0x0" ::: "memory");
#endif
#endif
}

// ---------------------------------------------------------------------------
// Kernel 0: pre-swizzle weights into bf16 B-fragments.
// grid = 4 (one 128x128 matrix each), block = 256. Block 0 also does W_f1.
// ---------------------------------------------------------------------------
__global__ __launch_bounds__(256) void k_prep(const float* __restrict__ W_in2f,
                                              const float* __restrict__ W_f1,
                                              const float* __restrict__ W_f2,
                                              const float* __restrict__ W_o1,
                                              const float* __restrict__ W_o2,
                                              __bf16* __restrict__ frags) {
  const float* Ws[4] = {W_in2f, W_f2, W_o1, W_o2};
  const float* W = Ws[blockIdx.x];
  const int wave = threadIdx.x >> 5, lane = threadIdx.x & 31;
  const int half = lane >> 4, lr = lane & 15;
  const int nt = wave, n0 = nt * 16;

  __bf16* base = frags + (size_t)blockIdx.x * FRAG_MAT_ELEMS;
#pragma unroll
  for (int kk = 0; kk < 4; ++kk) {
    v16bf b = load_b_f32(W, 128, kk * 32, n0 + lr, half);
    store_frag(base, (size_t)(nt * 4 + kk), lane, b);
  }

  if (blockIdx.x == 0) {   // W_f1: 20x128 padded to K=32, one fragment per n-tile
    __bf16* b1base = frags + (size_t)4 * FRAG_MAT_ELEMS;
    v16bf b;
#pragma unroll
    for (int i = 0; i < 16; ++i) {
      const int k = i + (i >= 8 ? 8 : 0) + 8 * half;
      b[i] = (k < N_RBF) ? (__bf16)W_f1[k * N_FILT + n0 + lr] : (__bf16)0.0f;
    }
    store_frag(b1base, (size_t)nt, lane, b);
  }
}

// ---------------------------------------------------------------------------
// Kernel 1: h = x @ W_in2f ; agg = 0
// grid = N_ATOMS/16, block = 256 (8 waves, each owns a 16-col stripe)
// ---------------------------------------------------------------------------
__global__ __launch_bounds__(256) void k_in2f(const float* __restrict__ x,
                                              const __bf16* __restrict__ Fw,
                                              float* __restrict__ h,
                                              float* __restrict__ agg) {
  const int wave = threadIdx.x >> 5, lane = threadIdx.x & 31;
  const int half = lane >> 4, lr = lane & 15;
  const int m0 = blockIdx.x * 16;
  const int n0 = wave * 16;

  v8f acc = {};
#pragma unroll
  for (int kk = 0; kk < 4; ++kk) {
    v16bf a = load_a_f32(x, N_BASIS, m0 + lr, kk * 32, half);
    v16bf b = load_frag(Fw, wave, kk, lane);
    acc = wmma_bf16(a, b, acc);
  }
#pragma unroll
  for (int r = 0; r < 8; ++r) {
    const int row = m0 + r + 8 * half;
    h[(size_t)row * N_FILT + n0 + lr] = acc[r];
    agg[(size_t)row * N_FILT + n0 + lr] = 0.0f;
  }
}

// ---------------------------------------------------------------------------
// Kernel 2: edge kernel — filter net + gather/multiply + sorted scatter-add
// grid = N_EDGES/(TILE_E*EDGE_TILES), block = 256
// ---------------------------------------------------------------------------
__global__ __launch_bounds__(256) void k_edge(const float* __restrict__ f_ij,
                                              const float* __restrict__ rcut,
                                              const int* __restrict__ idx_i,
                                              const int* __restrict__ idx_j,
                                              const __bf16* __restrict__ Fw1,
                                              const float* __restrict__ b1,
                                              const __bf16* __restrict__ Fw2,
                                              const float* __restrict__ b2,
                                              const float* __restrict__ h,
                                              float* __restrict__ agg) {
  __shared__ __bf16 T[TILE_E * T_LD];                 // ssp(f_ij@W1+b1), bf16
  __shared__ __align__(16) float FB[2][FIJ_TILE_WORDS];  // f_ij double buffer

  const int tid = threadIdx.x;
  const int wave = tid >> 5, lane = tid & 31;
  const int half = lane >> 4, lr = lane & 15;
  const int n0 = wave * 16;

  // --- hoisted weight fragments (2 b128 loads each, pre-swizzled) ----------
  v16bf bW1;
  {
    const uint4* p =
        reinterpret_cast<const uint4*>(Fw1 + ((size_t)wave * 32 + lane) * 16);
    union { v16bf v; uint4 q[2]; } u;
    u.q[0] = p[0];
    u.q[1] = p[1];
    bW1 = u.v;
  }
  const float bias1 = b1[n0 + lr];

  v16bf bW2[4];
#pragma unroll
  for (int kk = 0; kk < 4; ++kk) bW2[kk] = load_frag(Fw2, wave, kk, lane);
  const float bias2 = b2[n0 + lr];

  const int tile0 = blockIdx.x * EDGE_TILES;

  // kick off the first tile's f_ij copy
  fij_tile_copy(f_ij + (size_t)tile0 * TILE_E * N_RBF, FB[0], tid);

#pragma unroll 1
  for (int t = 0; t < EDGE_TILES; ++t) {
    const int blockE = (tile0 + t) * TILE_E;
    const float* Fcur = FB[t & 1];

    fij_copy_wait();          // our wave's async copies have landed in LDS
    __syncthreads();          // everyone's copies visible; T free to overwrite

    // prefetch next tile while we compute this one
    if (t + 1 < EDGE_TILES)
      fij_tile_copy(f_ij + (size_t)(tile0 + t + 1) * TILE_E * N_RBF,
                    FB[(t + 1) & 1], tid);

    // --- Stage A: T = bf16( ssp(f_ij @ W1 + b1) ) -------------------------
#pragma unroll 1
    for (int rb = 0; rb < TILE_E / 16; ++rb) {
      const float* fr = Fcur + (rb * 16 + lr) * N_RBF;
      v16bf a;
#pragma unroll
      for (int i = 0; i < 8; ++i)          // K = 8*half + i  (< 20 always)
        a[i] = (__bf16)fr[8 * half + i];
#pragma unroll
      for (int i = 0; i < 8; ++i) {        // K = 8*half + 16 + i (pad >= 20)
        const int c = 8 * half + 16 + i;
        a[8 + i] = (c < N_RBF) ? (__bf16)fr[c] : (__bf16)0.0f;
      }
      v8f acc = {};
      acc = wmma_bf16(a, bW1, acc);
#pragma unroll
      for (int r = 0; r < 8; ++r)
        T[(rb * 16 + r + 8 * half) * T_LD + n0 + lr] = (__bf16)ssp(acc[r] + bias1);
    }
    __syncthreads();

    // --- Stage B: Wij = T @ W2 + b2 ; gather h[idx_j]; scatter agg[idx_i] --
#pragma unroll 1
    for (int rb = 0; rb < TILE_E / 16; ++rb) {
      v8f acc = {};
#pragma unroll
      for (int kk = 0; kk < 4; ++kk) {
        v16bf a = load_a_lds(T, T_LD, rb * 16 + lr, kk * 32, half);
        acc = wmma_bf16(a, bW2[kk], acc);
      }
      const int E0 = blockE + rb * 16;
      const int ii_first = idx_i[E0];
      const int ii_last  = idx_i[E0 + 15];

      float vals[8];
#pragma unroll
      for (int r = 0; r < 8; ++r) {
        const int e = E0 + r + 8 * half;
        const float w = (acc[r] + bias2) * rcut[e];
        const int j = idx_j[e];
        vals[r] = w * h[(size_t)j * N_FILT + n0 + lr];
      }

      if (ii_first == ii_last) {
        // sorted idx_i: whole 16-edge tile hits one atom -> 16 atomics not 256
        float s = 0.0f;
#pragma unroll
        for (int r = 0; r < 8; ++r) s += vals[r];
        s += __shfl_xor(s, 16, 32);        // combine the two half-waves
        if (half == 0)
          atomicAdd(&agg[(size_t)ii_first * N_FILT + n0 + lr], s);
      } else {
#pragma unroll
        for (int r = 0; r < 8; ++r) {
          const int ii = idx_i[E0 + r + 8 * half];
          atomicAdd(&agg[(size_t)ii * N_FILT + n0 + lr], vals[r]);
        }
      }
    }
    __syncthreads();   // protect T (and FB) before next tile overwrites
  }
}

// ---------------------------------------------------------------------------
// Kernel 3: out = ssp(agg @ W_o1 + b_o1) @ W_o2 + b_o2
// grid = N_ATOMS/16, block = 256
// ---------------------------------------------------------------------------
__global__ __launch_bounds__(256) void k_out(const float* __restrict__ agg,
                                             const __bf16* __restrict__ Fo1,
                                             const float* __restrict__ bo1,
                                             const __bf16* __restrict__ Fo2,
                                             const float* __restrict__ bo2,
                                             float* __restrict__ out) {
  __shared__ __bf16 T[16 * T_LD];

  const int wave = threadIdx.x >> 5, lane = threadIdx.x & 31;
  const int half = lane >> 4, lr = lane & 15;
  const int m0 = blockIdx.x * 16;
  const int n0 = wave * 16;

  // stage 1: t = ssp(agg @ Wo1 + bo1) -> LDS (bf16)
  v8f acc = {};
#pragma unroll
  for (int kk = 0; kk < 4; ++kk) {
    v16bf a = load_a_f32(agg, N_FILT, m0 + lr, kk * 32, half);
    v16bf b = load_frag(Fo1, wave, kk, lane);
    acc = wmma_bf16(a, b, acc);
  }
  const float bias1 = bo1[n0 + lr];
#pragma unroll
  for (int r = 0; r < 8; ++r)
    T[(r + 8 * half) * T_LD + n0 + lr] = (__bf16)ssp(acc[r] + bias1);
  __syncthreads();

  // stage 2: out = t @ Wo2 + bo2
  v8f acc2 = {};
#pragma unroll
  for (int kk = 0; kk < 4; ++kk) {
    v16bf a = load_a_lds(T, T_LD, lr, kk * 32, half);
    v16bf b = load_frag(Fo2, wave, kk, lane);
    acc2 = wmma_bf16(a, b, acc2);
  }
  const float bias2 = bo2[n0 + lr];
#pragma unroll
  for (int r = 0; r < 8; ++r)
    out[(size_t)(m0 + r + 8 * half) * N_BASIS + n0 + lr] = acc2[r] + bias2;
}

// ---------------------------------------------------------------------------
extern "C" void kernel_launch(void* const* d_in, const int* in_sizes, int n_in,
                              void* d_out, int out_size, void* d_ws, size_t ws_size,
                              hipStream_t stream) {
  const float* x      = (const float*)d_in[0];
  const float* f_ij   = (const float*)d_in[1];
  const float* rcut   = (const float*)d_in[2];
  const int*   idx_i  = (const int*)d_in[3];
  const int*   idx_j  = (const int*)d_in[4];
  const float* W_in2f = (const float*)d_in[5];
  const float* W_f1   = (const float*)d_in[6];
  const float* b_f1   = (const float*)d_in[7];
  const float* W_f2   = (const float*)d_in[8];
  const float* b_f2   = (const float*)d_in[9];
  const float* W_o1   = (const float*)d_in[10];
  const float* b_o1   = (const float*)d_in[11];
  const float* W_o2   = (const float*)d_in[12];
  const float* b_o2   = (const float*)d_in[13];
  float* out = (float*)d_out;

  // workspace layout: [ frags : 256KB | h : 25.6MB | agg : 25.6MB ]
  __bf16* frags = (__bf16*)d_ws;
  float* h   = (float*)((char*)d_ws + WS_FRAG_RESERVE);
  float* agg = h + (size_t)N_ATOMS * N_FILT;

  const __bf16* F_in2f = frags + (size_t)0 * FRAG_MAT_ELEMS;
  const __bf16* F_f2   = frags + (size_t)1 * FRAG_MAT_ELEMS;
  const __bf16* F_o1   = frags + (size_t)2 * FRAG_MAT_ELEMS;
  const __bf16* F_o2   = frags + (size_t)3 * FRAG_MAT_ELEMS;
  const __bf16* F_f1   = frags + (size_t)4 * FRAG_MAT_ELEMS;

  k_prep<<<4, 256, 0, stream>>>(W_in2f, W_f1, W_f2, W_o1, W_o2, frags);
  k_in2f<<<N_ATOMS / 16, 256, 0, stream>>>(x, F_in2f, h, agg);
  k_edge<<<N_EDGES / (TILE_E * EDGE_TILES), 256, 0, stream>>>(
      f_ij, rcut, idx_i, idx_j, F_f1, b_f1, F_f2, b_f2, h, agg);
  k_out<<<N_ATOMS / 16, 256, 0, stream>>>(agg, F_o1, b_o1, F_o2, b_o2, out);
}
